// MemoryGateBlock_14534169330265
// MI455X (gfx1250) — compile-verified, matrix operand order checked by reading, not run on hardware
//
#include <hip/hip_runtime.h>

// ---- problem constants (match reference) ----
#define Bb 32768
#define Mm 10
#define Dd 256
#define Hh 256
#define NHh 8
#define HDd 32
#define BB 8                // batches per workgroup
#define ROWS (BB*Mm)        // 80 GRU rows = 5 x 16 tiles, exact
#define RT 5                // row tiles
#define THREADS 256         // 8 wave32s

typedef __bf16 bf16_t;
typedef __attribute__((ext_vector_type(16))) __bf16 bfx16;
typedef __attribute__((ext_vector_type(8)))  __bf16 bfx8;
typedef __attribute__((ext_vector_type(8)))  float  fx8;

__device__ __forceinline__ bf16_t f2bf(float f) {
  unsigned u = __builtin_bit_cast(unsigned, f);
  unsigned r = u + 0x7FFFu + ((u >> 16) & 1u);   // round-to-nearest-even
  unsigned short h = (unsigned short)(r >> 16);
  return __builtin_bit_cast(bf16_t, h);
}
__device__ __forceinline__ float bf2f(bf16_t b) {
  unsigned u = ((unsigned)__builtin_bit_cast(unsigned short, b)) << 16;
  return __builtin_bit_cast(float, u);
}
__device__ __forceinline__ fx8 wmma_bf16(bfx16 a, bfx16 b, fx8 c) {
  return __builtin_amdgcn_wmma_f32_16x16x32_bf16(false, a, false, b, (short)0, c, false, false);
}
__device__ __forceinline__ bfx16 cat16(bfx8 lo, bfx8 hi) {
  return __builtin_shufflevector(lo, hi, 0,1,2,3,4,5,6,7,8,9,10,11,12,13,14,15);
}

// A-fragment (16x32 bf16) from row-major bf16 in LDS: lane (l%16 = row M,
// l/16 = K-half) needs K values kOff+8g+{0..7} and +{16..23}: two 16B runs.
__device__ __forceinline__ bfx16 fragA_lds(const bf16_t* __restrict__ p) {
  bfx8 lo = *(const bfx8*)(p);
  bfx8 hi = *(const bfx8*)(p + 16);
  return cat16(lo, hi);
}

// B-fragment from pre-packed bf16 weights in d_ws: one 32B load per lane.
__device__ __forceinline__ bfx16 fragB_packed(const bf16_t* __restrict__ P,
                                              int KC, int kc, int ct, int lane) {
  const bf16_t* p = P + ((size_t)((ct*KC + kc)*32 + lane) << 4);
  bfx8 lo = *(const bfx8*)(p);
  bfx8 hi = *(const bfx8*)(p + 8);
  return cat16(lo, hi);
}

// Fallback B-fragment straight from fp32 global weights (ws too small).
__device__ __forceinline__ bfx16 fragB_f32(const float* __restrict__ W, int ldw,
                                           int kOff, int nglob, int g) {
  const float* p = W + (size_t)(kOff + 8*g) * ldw + nglob;
  bfx16 b;
#pragma unroll
  for (int v = 0; v < 8; ++v) {
    const int kb = (v < 4) ? (2*v) : (8 + 2*v);
    b[2*v]   = f2bf(p[(size_t)kb * ldw]);
    b[2*v+1] = f2bf(p[(size_t)(kb+1) * ldw]);
  }
  return b;
}

template <bool PACKED>
__device__ __forceinline__ bfx16 getB(const bf16_t* P, const float* W, int ldw,
                                      int KC, int kc, int ct, int lane,
                                      int nglob, int gg) {
  if constexpr (PACKED) return fragB_packed(P, KC, kc, ct, lane);
  else                  return fragB_f32(W, ldw, kc*32, nglob, gg);
}

// async DMA: global (16B) -> LDS, tracked by ASYNCcnt
__device__ __forceinline__ void async_g2l_b128(unsigned lds_byte_off,
                                               const void* gaddr) {
  unsigned long long ga = (unsigned long long)gaddr;
  asm volatile("global_load_async_to_lds_b128 %0, %1, off"
               :: "v"(lds_byte_off), "v"(ga) : "memory");
}
__device__ __forceinline__ void wait_async0() {
  asm volatile("s_wait_asynccnt 0x0" ::: "memory");
}

// ---- one-shot weight repack: fp32 row-major [K][N] -> bf16 fragment order ----
__global__ void pack_weights(const float* __restrict__ W, bf16_t* __restrict__ P,
                             int K, int N) {
  const int idx = blockIdx.x * 256 + threadIdx.x;
  if (idx >= K * N) return;
  const int e     = idx & 15;          // element within lane's fragment
  const int lane  = (idx >> 4) & 31;
  const int chunk = idx >> 9;          // ct*KC + kc
  const int KC = K >> 5;
  const int kc = chunk % KC, ct = chunk / KC;
  const int g = lane >> 4, n = ct*16 + (lane & 15);
  const int v = e >> 1;
  const int kb = ((v < 4) ? (2*v) : (8 + 2*v)) + (e & 1);
  const int k = kc*32 + 8*g + kb;
  P[idx] = f2bf(W[(size_t)k * N + n]);
}

template <bool PACKED>
__global__ __launch_bounds__(THREADS, 1)
void memgate_fused(const float* __restrict__ cur_g, const float* __restrict__ prev_g,
                   const float* __restrict__ Wz, const float* __restrict__ bz,
                   const float* __restrict__ Wr, const float* __restrict__ br,
                   const float* __restrict__ Wh, const float* __restrict__ bh,
                   const float* __restrict__ Wq, const float* __restrict__ bq,
                   const float* __restrict__ Wk, const float* __restrict__ bk,
                   const float* __restrict__ Wo, const float* __restrict__ bo,
                   const float* __restrict__ g2, const float* __restrict__ be2,
                   const bf16_t* __restrict__ Pz, const bf16_t* __restrict__ Pr,
                   const bf16_t* __restrict__ Ph, const bf16_t* __restrict__ Pq,
                   const bf16_t* __restrict__ Pk, const bf16_t* __restrict__ Po,
                   float* __restrict__ out_fused, float* __restrict__ out_upd,
                   float* __restrict__ out_attn) {
  extern __shared__ char smem[];
  float*  s_prev  = (float*)smem;                   // [80][256] fp32 prev
  float*  s_zk    = s_prev + ROWS*Hh;               // [80][256] cur stage -> z -> k
  float*  s_q     = s_zk   + ROWS*Hh;               // [8][256]  q
  float*  s_fus   = s_q    + BB*Hh;                 // [8][256]  fused pre-LN
  float*  s_anm   = s_fus  + BB*Dd;                 // [8][8][10] per-head attn
  float*  s_attn  = s_anm  + BB*NHh*Mm;             // [8][10]
  bf16_t* s_prevb = (bf16_t*)(s_attn + ROWS);       // [80][256] bf16 prev
  bf16_t* s_rp    = s_prevb + ROWS*Hh;              // [80][256] bf16 r*prev
  bf16_t* s_ub    = s_rp   + ROWS*Hh;               // [80][256] bf16 upd
  bf16_t* s_curb  = s_ub   + ROWS*Hh;               // [8][256]  bf16 current
  bf16_t* s_wmb   = s_curb + BB*Dd;                 // [8][256]  bf16 weighted mem

  const int tid  = threadIdx.x;
  const int wave = tid >> 5;
  const int lane = tid & 31;
  const int gg   = lane >> 4;
  const int nl   = lane & 15;
  const int b0   = blockIdx.x * BB;

  // ---- stage inputs via async DMA straight into LDS (ASYNCcnt) ----
  {
    const float4* pg = (const float4*)(prev_g + (size_t)b0 * Mm * Hh);
#pragma unroll 1
    for (int i = tid; i < (ROWS*Hh)/4; i += THREADS)
      async_g2l_b128((unsigned)((char*)(s_prev + i*4) - smem), pg + i);
    const float4* cg = (const float4*)(cur_g + (size_t)b0 * Dd);
#pragma unroll 1
    for (int i = tid; i < (BB*Dd)/4; i += THREADS)   // cur fp32 -> s_zk scratch
      async_g2l_b128((unsigned)((char*)(s_zk + i*4) - smem), cg + i);
    wait_async0();
  }
  __syncthreads();
  // bf16 shadow copies (from LDS)
  for (int i = tid; i < BB*Dd; i += THREADS)
    s_curb[i] = f2bf(s_zk[i]);
  {
    const float4* ps = (const float4*)s_prev;
    for (int i = tid; i < (ROWS*Hh)/4; i += THREADS) {
      float4 v = ps[i];
      const int j = i * 4;
      s_prevb[j]   = f2bf(v.x);
      s_prevb[j+1] = f2bf(v.y);
      s_prevb[j+2] = f2bf(v.z);
      s_prevb[j+3] = f2bf(v.w);
    }
  }
  __syncthreads();

  // hoisted per-row-tile A base pointers (lane's A row = nl within each tile)
  const bf16_t* curA[RT];
  const bf16_t* prevA[RT];
  const bf16_t* rpA[RT];
  const bf16_t* ubA[RT];
#pragma unroll
  for (int rt = 0; rt < RT; ++rt) {
    const int row = rt*16 + nl;
    curA[rt]  = s_curb  + (row / Mm) * Dd + 8*gg;
    prevA[rt] = s_prevb + row * Hh + 8*gg;
    rpA[rt]   = s_rp    + row * Hh + 8*gg;
    ubA[rt]   = s_ub    + row * Hh + 8*gg;
  }
  const bf16_t* qA  = s_curb + (nl & 7) * Dd + 8*gg;
  const bf16_t* wmA = s_wmb  + (nl & 7) * Hh + 8*gg;

  // ---------------- phase 1: z and r gates (K=512) ----------------
  for (int c = 0; c < 2; ++c) {
    const int ct = wave*2 + c;
    const int nglob = ct*16 + nl;
    fx8 accz[RT] = {};
    fx8 accr[RT] = {};
#pragma unroll 1
    for (int kc = 0; kc < 8; ++kc) {          // first half: current features
      bfx16 fz = getB<PACKED>(Pz, Wz, Hh, 16, kc, ct, lane, nglob, gg);
      bfx16 fr = getB<PACKED>(Pr, Wr, Hh, 16, kc, ct, lane, nglob, gg);
#pragma unroll
      for (int rt = 0; rt < RT; ++rt) {
        bfx16 fa = fragA_lds(curA[rt] + kc*32);
        accz[rt] = wmma_bf16(fa, fz, accz[rt]);
        accr[rt] = wmma_bf16(fa, fr, accr[rt]);
      }
    }
#pragma unroll 1
    for (int kc = 8; kc < 16; ++kc) {         // second half: prev memory
      bfx16 fz = getB<PACKED>(Pz, Wz, Hh, 16, kc, ct, lane, nglob, gg);
      bfx16 fr = getB<PACKED>(Pr, Wr, Hh, 16, kc, ct, lane, nglob, gg);
#pragma unroll
      for (int rt = 0; rt < RT; ++rt) {
        bfx16 fa = fragA_lds(prevA[rt] + (kc-8)*32);
        accz[rt] = wmma_bf16(fa, fz, accz[rt]);
        accr[rt] = wmma_bf16(fa, fr, accr[rt]);
      }
    }
    const float bzv = bz[nglob], brv = br[nglob];
#pragma unroll
    for (int rt = 0; rt < RT; ++rt) {
#pragma unroll
      for (int v = 0; v < 8; ++v) {
        const int row = rt*16 + v + 8*gg;      // C/D layout: M = v + 8*(l/16)
        float zf = 1.f / (1.f + __expf(-(accz[rt][v] + bzv)));
        float rf = 1.f / (1.f + __expf(-(accr[rt][v] + brv)));
        s_zk[row*Hh + nglob] = zf;
        s_rp[row*Hh + nglob] = f2bf(rf * s_prev[row*Hh + nglob]);
      }
    }
  }
  __syncthreads();

  // ---------------- phase 2: candidate + GRU blend ----------------
  for (int c = 0; c < 2; ++c) {
    const int ct = wave*2 + c;
    const int nglob = ct*16 + nl;
    fx8 acch[RT] = {};
#pragma unroll 1
    for (int kc = 0; kc < 8; ++kc) {
      bfx16 fw = getB<PACKED>(Ph, Wh, Hh, 16, kc, ct, lane, nglob, gg);
#pragma unroll
      for (int rt = 0; rt < RT; ++rt)
        acch[rt] = wmma_bf16(fragA_lds(curA[rt] + kc*32), fw, acch[rt]);
    }
#pragma unroll 1
    for (int kc = 8; kc < 16; ++kc) {
      bfx16 fw = getB<PACKED>(Ph, Wh, Hh, 16, kc, ct, lane, nglob, gg);
#pragma unroll
      for (int rt = 0; rt < RT; ++rt)
        acch[rt] = wmma_bf16(fragA_lds(rpA[rt] + (kc-8)*32), fw, acch[rt]);
    }
    const float bhv = bh[nglob];
#pragma unroll
    for (int rt = 0; rt < RT; ++rt) {
#pragma unroll
      for (int v = 0; v < 8; ++v) {
        const int row = rt*16 + v + 8*gg;
        float cd = tanhf(acch[rt][v] + bhv);
        float zf = s_zk[row*Hh + nglob];
        float pv = s_prev[row*Hh + nglob];
        float u  = (1.f - zf) * pv + zf * cd;
        s_ub[row*Hh + nglob] = f2bf(u);
        out_upd[((size_t)b0*Mm + row) * Hh + nglob] = u;
      }
    }
  }
  __syncthreads();

  // ---------------- phase 3: k = upd @ Wk (reuse s_zk), q = cur @ Wq ----------------
  for (int c = 0; c < 2; ++c) {
    const int ct = wave*2 + c;
    const int nglob = ct*16 + nl;
    fx8 acck[RT] = {};
#pragma unroll 1
    for (int kc = 0; kc < 8; ++kc) {
      bfx16 fw = getB<PACKED>(Pk, Wk, Hh, 8, kc, ct, lane, nglob, gg);
#pragma unroll
      for (int rt = 0; rt < RT; ++rt)
        acck[rt] = wmma_bf16(fragA_lds(ubA[rt] + kc*32), fw, acck[rt]);
    }
    const float bkv = bk[nglob];
#pragma unroll
    for (int rt = 0; rt < RT; ++rt)
#pragma unroll
      for (int v = 0; v < 8; ++v) {
        const int row = rt*16 + v + 8*gg;
        s_zk[row*Hh + nglob] = acck[rt][v] + bkv;   // k values
      }

    fx8 accq = {};
#pragma unroll 1
    for (int kc = 0; kc < 8; ++kc) {
      bfx16 fw = getB<PACKED>(Pq, Wq, Hh, 8, kc, ct, lane, nglob, gg);
      accq = wmma_bf16(fragA_lds(qA + kc*32), fw, accq);
    }
    const float bqv = bq[nglob];
    if (gg == 0) {
#pragma unroll
      for (int v = 0; v < 8; ++v)
        s_q[v*Hh + nglob] = accq[v] + bqv;          // batch rows 0..7
    }
  }
  __syncthreads();

  // ---------------- phase 4: scores -> softmax -> head mean ----------------
  if (tid < BB*NHh) {                   // 64 (batch, head) pairs
    const int b = tid >> 3, n = tid & 7;
    const float scale = 0.17677669529663687f;   // 1/sqrt(32)
    float sc[Mm];
    float mx = -3.4e38f;
    for (int m = 0; m < Mm; ++m) {
      const float* qp = s_q  +  b*Hh + n*HDd;
      const float* kp = s_zk + (b*Mm + m)*Hh + n*HDd;
      float d = 0.f;
#pragma unroll
      for (int t = 0; t < HDd; ++t) d += qp[t] * kp[t];
      sc[m] = d * scale;
      mx = fmaxf(mx, sc[m]);
    }
    float ssum = 0.f;
    for (int m = 0; m < Mm; ++m) { sc[m] = __expf(sc[m] - mx); ssum += sc[m]; }
    const float inv = 1.f / ssum;
    for (int m = 0; m < Mm; ++m) s_anm[(b*NHh + n)*Mm + m] = sc[m] * inv;
  }
  __syncthreads();
  if (tid < ROWS) {                     // 80 (batch, m) pairs
    const int b = tid / Mm, m = tid % Mm;
    float a = 0.f;
    for (int n = 0; n < NHh; ++n) a += s_anm[(b*NHh + n)*Mm + m];
    a *= (1.f / NHh);
    s_attn[tid] = a;
    out_attn[(size_t)(b0 + b)*Mm + m] = a;
  }
  __syncthreads();

  // ---------------- phase 5: weighted memory (bf16 for fused GEMM) ----------------
  for (int i = tid; i < BB*Hh; i += THREADS) {
    const int b = i >> 8, h = i & 255;
    float a = 0.f;
    for (int m = 0; m < Mm; ++m)
      a += s_attn[b*Mm + m] * bf2f(s_ub[(b*Mm + m)*Hh + h]);
    s_wmb[i] = f2bf(a);
  }
  __syncthreads();

  // ---------------- phase 6: fused = [cur | wm] @ Wo (K=512) ----------------
  for (int c = 0; c < 2; ++c) {
    const int ct = wave*2 + c;
    const int nglob = ct*16 + nl;
    fx8 accf = {};
#pragma unroll 1
    for (int kc = 0; kc < 8; ++kc) {
      bfx16 fw = getB<PACKED>(Po, Wo, Dd, 16, kc, ct, lane, nglob, gg);
      accf = wmma_bf16(fragA_lds(qA + kc*32), fw, accf);
    }
#pragma unroll 1
    for (int kc = 8; kc < 16; ++kc) {
      bfx16 fw = getB<PACKED>(Po, Wo, Dd, 16, kc, ct, lane, nglob, gg);
      accf = wmma_bf16(fragA_lds(wmA + (kc-8)*32), fw, accf);
    }
    const float bov = bo[nglob];
    if (gg == 0) {
#pragma unroll
      for (int v = 0; v < 8; ++v)
        s_fus[v*Dd + nglob] = accf[v] + bov;
    }
  }
  __syncthreads();

  // ---------------- phase 7: LayerNorm (one wave per batch row) ----------------
  {
    const int b = wave;                 // 8 waves, batches 0..7
    float x[8];
    float lsum = 0.f;
#pragma unroll
    for (int i = 0; i < 8; ++i) { x[i] = s_fus[b*Dd + lane + 32*i]; lsum += x[i]; }
    for (int off = 16; off > 0; off >>= 1) lsum += __shfl_xor(lsum, off, 32);
    const float mu = lsum * (1.f / Dd);
    float lsq = 0.f;
#pragma unroll
    for (int i = 0; i < 8; ++i) { float d = x[i] - mu; lsq += d * d; }
    for (int off = 16; off > 0; off >>= 1) lsq += __shfl_xor(lsq, off, 32);
    const float inv = rsqrtf(lsq * (1.f / Dd) + 1e-5f);
#pragma unroll
    for (int i = 0; i < 8; ++i) {
      const int dcol = lane + 32*i;
      out_fused[(size_t)(b0 + b)*Dd + dcol] = (x[i] - mu) * inv * g2[dcol] + be2[dcol];
    }
  }
}

extern "C" void kernel_launch(void* const* d_in, const int* in_sizes, int n_in,
                              void* d_out, int out_size, void* d_ws, size_t ws_size,
                              hipStream_t stream) {
  (void)in_sizes; (void)n_in; (void)out_size;
  const float* cur  = (const float*)d_in[0];
  const float* prev = (const float*)d_in[1];
  const float* Wz   = (const float*)d_in[2];
  const float* bz   = (const float*)d_in[3];
  const float* Wr   = (const float*)d_in[4];
  const float* br   = (const float*)d_in[5];
  const float* Wh   = (const float*)d_in[6];
  const float* bh   = (const float*)d_in[7];
  const float* Wq   = (const float*)d_in[8];
  const float* bq   = (const float*)d_in[9];
  const float* Wk   = (const float*)d_in[10];
  const float* bk   = (const float*)d_in[11];
  const float* Wo   = (const float*)d_in[12];
  const float* bo   = (const float*)d_in[13];
  const float* g2   = (const float*)d_in[14];
  const float* be2  = (const float*)d_in[15];

  float* out       = (float*)d_out;
  float* out_fused = out;                               // [B, D]
  float* out_upd   = out + (size_t)Bb * Dd;             // [B, M, H]
  float* out_attn  = out_upd + (size_t)Bb * Mm * Hh;    // [B, M]

  // packed bf16 weight layout in workspace
  const size_t nZ = (size_t)(Dd+Hh) * Hh;   // 131072 (Wz/Wr/Wh/Wo)
  const size_t nK = (size_t)Hh * Hh;        // 65536  (Wq/Wk)
  bf16_t* Pz = (bf16_t*)d_ws;
  bf16_t* Pr = Pz + nZ;
  bf16_t* Ph = Pr + nZ;
  bf16_t* Pk = Ph + nZ;
  bf16_t* Pq = Pk + nK;
  bf16_t* Po = Pq + nK;
  const size_t ws_needed = (4*nZ + 2*nK) * sizeof(bf16_t);  // 1.31 MB
  const bool packed = (ws_size >= ws_needed) && (d_ws != nullptr);

  if (packed) {
    pack_weights<<<(nZ + 255)/256, 256, 0, stream>>>(Wz, Pz, Dd+Hh, Hh);
    pack_weights<<<(nZ + 255)/256, 256, 0, stream>>>(Wr, Pr, Dd+Hh, Hh);
    pack_weights<<<(nZ + 255)/256, 256, 0, stream>>>(Wh, Ph, Dd+Hh, Hh);
    pack_weights<<<(nK + 255)/256, 256, 0, stream>>>(Wk, Pk, Hh, Hh);
    pack_weights<<<(nK + 255)/256, 256, 0, stream>>>(Wq, Pq, Hh, Hh);
    pack_weights<<<(nZ + 255)/256, 256, 0, stream>>>(Wo, Po, Dd+Hh, Dd);
  }

  // dynamic LDS = 314,176 bytes (<= 320 KB/WGP)
  const size_t smem_floats = (size_t)ROWS*Hh*2 + (size_t)BB*Hh + (size_t)BB*Dd
                           + (size_t)BB*NHh*Mm + ROWS;
  const size_t smem_bytes  = smem_floats*4
                           + ((size_t)ROWS*Hh*3 + (size_t)BB*Dd*2)*2;

  dim3 grid(Bb / BB), block(THREADS);
  if (packed) {
    hipLaunchKernelGGL(HIP_KERNEL_NAME(memgate_fused<true>), grid, block, smem_bytes, stream,
                       cur, prev, Wz, bz, Wr, br, Wh, bh, Wq, bq, Wk, bk,
                       Wo, bo, g2, be2, Pz, Pr, Ph, Pq, Pk, Po,
                       out_fused, out_upd, out_attn);
  } else {
    hipLaunchKernelGGL(HIP_KERNEL_NAME(memgate_fused<false>), grid, block, smem_bytes, stream,
                       cur, prev, Wz, bz, Wr, br, Wh, bh, Wq, bq, Wk, bk,
                       Wo, bo, g2, be2, Pz, Pr, Ph, Pq, Pk, Po,
                       out_fused, out_upd, out_attn);
  }
}
